// GSpadeModel_21277267984970
// MI455X (gfx1250) — matrix operand order repaired and checked by reading, not compiled, not run on hardware
//
#include <hip/hip_runtime.h>
#include <hip/hip_bf16.h>
#include <math.h>

// ---------------------------------------------------------------------------
// Types for CDNA5 WMMA (wave32): v_wmma_f32_16x16x32_bf16
// ---------------------------------------------------------------------------
typedef float  v8f   __attribute__((ext_vector_type(8)));
typedef float  v4fv  __attribute__((ext_vector_type(4)));
typedef __bf16 v16bf __attribute__((ext_vector_type(16)));

#define NN     1024   // nodes
#define DM     512    // model dim
#define DH     256    // group dim
#define HL     256    // LSTM hidden
#define LSEQ   32
#define CHREL  20

__device__ __forceinline__ float gelu_erf(float x) {
  return 0.5f * x * (1.0f + erff(x * 0.70710678118654752f));
}
__device__ __forceinline__ float sigmoidf_(float x) {
  return 1.0f / (1.0f + __expf(-x));
}

// ---------------------------------------------------------------------------
// Generic WMMA GEMM:  C[m,n] = rs[m] * sum_k X[m,k]*B(k,n)  + bias[n] + add[m,n] (+ C)
//   bKmajor==0 : B is W[N,K] row-major (computes X @ W^T), b128 loads
//   bKmajor==1 : B is [K,N] row-major  (computes X @ B),   strided column loads
// One wave (32 threads) per 16x16 C tile. M,N mult of 16; K mult of 32.
// All buffers 256B-aligned; ld* multiples of 8 -> 16B-aligned vector loads.
// ---------------------------------------------------------------------------
__global__ void __launch_bounds__(32) gemm_wmma_kernel(
    const float* __restrict__ X, int ldx,
    const float* __restrict__ B, int ldb, int bKmajor,
    const float* __restrict__ bias,
    const float* __restrict__ rowscale,
    const float* __restrict__ addsrc, int ldadd,
    float* __restrict__ C, int ldc,
    int M, int N, int K, int accumulate)
{
  const int lane = threadIdx.x & 31;
  const int l15  = lane & 15;
  const int half = lane >> 4;
  const int tn = blockIdx.x * 16;
  const int tm = blockIdx.y * 16;

  v8f acc = {0.f,0.f,0.f,0.f,0.f,0.f,0.f,0.f};
  for (int k = 0; k < K; k += 32) {
    // A operand: lane row = tm+l15 ; halves hold K {0-15} | {16-31}
    const float* arow = X + (size_t)(tm + l15) * ldx + k + half * 16;
    __builtin_prefetch(arow + 64, 0, 0);
    const v4fv a0 = *(const v4fv*)(arow);
    const v4fv a1 = *(const v4fv*)(arow + 4);
    const v4fv a2 = *(const v4fv*)(arow + 8);
    const v4fv a3 = *(const v4fv*)(arow + 12);
    v16bf a;
#pragma unroll
    for (int e = 0; e < 4; ++e) {
      a[e]      = (__bf16)a0[e];
      a[e + 4]  = (__bf16)a1[e];
      a[e + 8]  = (__bf16)a2[e];
      a[e + 12] = (__bf16)a3[e];
    }
    // B operand: lane col = tn+l15 ; halves hold K {0-15} | {16-31}
    v16bf b;
    if (!bKmajor) {
      const float* brow = B + (size_t)(tn + l15) * ldb + k + half * 16;
      const v4fv b0 = *(const v4fv*)(brow);
      const v4fv b1 = *(const v4fv*)(brow + 4);
      const v4fv b2 = *(const v4fv*)(brow + 8);
      const v4fv b3 = *(const v4fv*)(brow + 12);
#pragma unroll
      for (int e = 0; e < 4; ++e) {
        b[e]      = (__bf16)b0[e];
        b[e + 4]  = (__bf16)b1[e];
        b[e + 8]  = (__bf16)b2[e];
        b[e + 12] = (__bf16)b3[e];
      }
    } else {
      const float* bcol = B + (size_t)(k + half * 16) * ldb + (tn + l15);
#pragma unroll
      for (int e = 0; e < 16; ++e) b[e] = (__bf16)bcol[(size_t)e * ldb];
    }
    acc = __builtin_amdgcn_wmma_f32_16x16x32_bf16(false, a, false, b,
                                                  (short)0, acc, false, false);
  }
#pragma unroll
  for (int r = 0; r < 8; ++r) {
    int row = tm + half * 8 + r;
    int col = tn + l15;
    float v = acc[r];
    if (rowscale) v *= rowscale[row];
    if (bias)     v += bias[col];
    if (addsrc)   v += addsrc[(size_t)row * ldadd + col];
    float* cp = C + (size_t)row * ldc + col;
    if (accumulate) v += *cp;
    *cp = v;
  }
}

// ---------------------------------------------------------------------------
// BiLSTM over char_feats, hidden-sum. One block = 16 sequences, one direction.
// 8 waves; wave w owns hidden columns [w*32, w*32+32) of all 4 gates.
// h state lives in LDS; recurrent GEMM h@Whh^T via WMMA (K=256 -> 8 steps).
// Writes hsum into cat[:, 512 + dir*256 + col].
// ---------------------------------------------------------------------------
__global__ void __launch_bounds__(256) bilstm_kernel(
    const float* __restrict__ xchar,                 // [1024,32,8]
    const float* __restrict__ Wih_f, const float* __restrict__ Whh_f, const float* __restrict__ b_f,
    const float* __restrict__ Wih_b, const float* __restrict__ Whh_b, const float* __restrict__ b_b,
    float* __restrict__ cat)                         // [1024,1536]
{
  const int dir = blockIdx.y;
  const float* Wih  = dir ? Wih_b : Wih_f;
  const float* Whh  = dir ? Whh_b : Whh_f;
  const float* bvec = dir ? b_b   : b_f;
  const int seq0 = blockIdx.x * 16;
  const int tid  = threadIdx.x;
  const int wave = tid >> 5;
  const int lane = tid & 31;
  const int l15  = lane & 15;
  const int half = lane >> 4;

  __shared__ float hbuf[16][HL];   // 16 KB current hidden state
  __shared__ float xs[16][8];

  for (int i = tid; i < 16 * HL; i += 256) ((float*)hbuf)[i] = 0.f;
  v8f cst[2]  = {{0.f,0.f,0.f,0.f,0.f,0.f,0.f,0.f},{0.f,0.f,0.f,0.f,0.f,0.f,0.f,0.f}};
  v8f hsum[2] = {{0.f,0.f,0.f,0.f,0.f,0.f,0.f,0.f},{0.f,0.f,0.f,0.f,0.f,0.f,0.f,0.f}};
  __syncthreads();

  for (int t = 0; t < LSEQ; ++t) {
    const int tt = dir ? (LSEQ - 1 - t) : t;
    if (tid < 128) {
      int r = tid >> 3, k = tid & 7;
      xs[r][k] = xchar[((size_t)(seq0 + r) * LSEQ + tt) * 8 + k];
    }
    __syncthreads();

    // z = b + x_t @ Wih^T  (K=8, scalar)  then += h @ Whh^T via WMMA
    v8f z[4][2];
#pragma unroll
    for (int gt = 0; gt < 4; ++gt)
#pragma unroll
      for (int s = 0; s < 2; ++s) {
        const int col = gt * HL + wave * 32 + s * 16 + l15;
        const float base = bvec[col];
        const float* wr = Wih + (size_t)col * 8;
#pragma unroll
        for (int r = 0; r < 8; ++r) {
          const int rl = half * 8 + r;
          float a = base;
#pragma unroll
          for (int k = 0; k < 8; ++k) a += xs[rl][k] * wr[k];
          z[gt][s][r] = a;
        }
      }

    for (int k = 0; k < HL; k += 32) {
      // A operand from LDS (ds_load_b128)
      const float* ar = &hbuf[l15][k + half * 16];
      const v4fv a0 = *(const v4fv*)(ar);
      const v4fv a1 = *(const v4fv*)(ar + 4);
      const v4fv a2 = *(const v4fv*)(ar + 8);
      const v4fv a3 = *(const v4fv*)(ar + 12);
      v16bf a;
#pragma unroll
      for (int e = 0; e < 4; ++e) {
        a[e]      = (__bf16)a0[e];
        a[e + 4]  = (__bf16)a1[e];
        a[e + 8]  = (__bf16)a2[e];
        a[e + 12] = (__bf16)a3[e];
      }
#pragma unroll
      for (int gt = 0; gt < 4; ++gt)
#pragma unroll
        for (int s = 0; s < 2; ++s) {
          const int col = gt * HL + wave * 32 + s * 16 + l15;
          const float* br = Whh + (size_t)col * HL + k + half * 16;
          const v4fv b0 = *(const v4fv*)(br);
          const v4fv b1 = *(const v4fv*)(br + 4);
          const v4fv b2 = *(const v4fv*)(br + 8);
          const v4fv b3 = *(const v4fv*)(br + 12);
          v16bf b;
#pragma unroll
          for (int e = 0; e < 4; ++e) {
            b[e]      = (__bf16)b0[e];
            b[e + 4]  = (__bf16)b1[e];
            b[e + 8]  = (__bf16)b2[e];
            b[e + 12] = (__bf16)b3[e];
          }
          z[gt][s] = __builtin_amdgcn_wmma_f32_16x16x32_bf16(false, a, false, b,
                                                             (short)0, z[gt][s], false, false);
        }
    }
    __syncthreads();   // all waves done reading hbuf

    // gates (torch order i,f,g,o): c = sig(f)*c + sig(i)*tanh(g); h = sig(o)*tanh(c)
#pragma unroll
    for (int s = 0; s < 2; ++s)
#pragma unroll
      for (int r = 0; r < 8; ++r) {
        float iv = sigmoidf_(z[0][s][r]);
        float fv = sigmoidf_(z[1][s][r]);
        float gv = tanhf(z[2][s][r]);
        float ov = sigmoidf_(z[3][s][r]);
        float c  = fv * cst[s][r] + iv * gv;
        float h  = ov * tanhf(c);
        cst[s][r]  = c;
        hsum[s][r] += h;
        hbuf[half * 8 + r][wave * 32 + s * 16 + l15] = h;
      }
    __syncthreads();   // new h visible before next step
  }

#pragma unroll
  for (int s = 0; s < 2; ++s)
#pragma unroll
    for (int r = 0; r < 8; ++r) {
      const int row = seq0 + half * 8 + r;
      const int col = 512 + dir * HL + wave * 32 + s * 16 + l15;
      cat[(size_t)row * 1536 + col] = hsum[s][r];
    }
}

// ---------------------------------------------------------------------------
// LayerNorm (rowwise) + optional residual + activation epilogue.
// act: 0=none 1=relu 2=gelu. One block per row, 256 threads, D in {256,512}.
// ---------------------------------------------------------------------------
__global__ void __launch_bounds__(256) ln_kernel(
    const float* __restrict__ in, int ldin,
    const float* __restrict__ res, int ldres,
    const float* __restrict__ gamma, const float* __restrict__ beta,
    float* __restrict__ out, int ldout, int D, int act)
{
  const int row = blockIdx.x;
  const int tid = threadIdx.x;
  const int nv  = D >> 8;          // 1 or 2
  __shared__ float red[256];

  float v[2] = {0.f, 0.f};
  float s = 0.f;
  for (int i = 0; i < nv; ++i) {
    const int c = tid + i * 256;
    float x = in[(size_t)row * ldin + c];
    if (res) x += res[(size_t)row * ldres + c];
    v[i] = x;
    s += x;
  }
  red[tid] = s; __syncthreads();
  for (int off = 128; off > 0; off >>= 1) {
    if (tid < off) red[tid] += red[tid + off];
    __syncthreads();
  }
  const float mean = red[0] / (float)D;
  __syncthreads();
  float sq = 0.f;
  for (int i = 0; i < nv; ++i) { float d = v[i] - mean; sq += d * d; }
  red[tid] = sq; __syncthreads();
  for (int off = 128; off > 0; off >>= 1) {
    if (tid < off) red[tid] += red[tid + off];
    __syncthreads();
  }
  const float rstd = rsqrtf(red[0] / (float)D + 1e-5f);
  for (int i = 0; i < nv; ++i) {
    const int c = tid + i * 256;
    float y = (v[i] - mean) * rstd * gamma[c] + beta[c];
    if (act == 1)      y = fmaxf(y, 0.f);
    else if (act == 2) y = gelu_erf(y);
    out[(size_t)row * ldout + c] = y;
  }
}

// ---------------------------------------------------------------------------
// Fold graph-builder into one 3x3x20->1 conv:
//  weff[0..179] = sum_k (sW[1,k]-sW[0,k]) * convK weights ; [180]=bias ; [181]=b1-b0
// ---------------------------------------------------------------------------
__global__ void prep_weff_kernel(
    const float* __restrict__ c0W, const float* __restrict__ c0b,
    const float* __restrict__ c1W, const float* __restrict__ c1b,
    const float* __restrict__ sW,  const float* __restrict__ sb,
    float* __restrict__ weff)
{
  const int tid = threadIdx.x;
  __shared__ float wd[16];
  if (tid < 16) wd[tid] = sW[16 + tid] - sW[tid];
  __syncthreads();
  if (tid < 180) {
    float a = 0.f;
    for (int k = 0; k < 16; ++k) {
      float w = (k < 8) ? c0W[k * 180 + tid] : c1W[(k - 8) * 180 + tid];
      a += wd[k] * w;
    }
    weff[tid] = a;
  } else if (tid == 180) {
    float a = 0.f;
    for (int k = 0; k < 16; ++k) a += wd[k] * ((k < 8) ? c0b[k] : c1b[k - 8]);
    weff[180] = a;
  } else if (tid == 181) {
    weff[181] = sb[1] - sb[0];
  }
}

// ---------------------------------------------------------------------------
// g[i,j] = 3x3x20 conv(rel)[i,j] + b_eff.  16x16 outputs per block.
// Interior blocks stage the 18x18 halo tile with CDNA5 async-to-LDS DMA
// (global_load_async_to_lds_b32, ASYNCcnt); boundary blocks use the
// predicated scalar path with zero padding.
// ---------------------------------------------------------------------------
__global__ void __launch_bounds__(256) conv_gb_kernel(
    const float* __restrict__ rel, const float* __restrict__ weff,
    float* __restrict__ g)
{
  __shared__ float sw[182];
  __shared__ float st[18][18];
  const int lin = threadIdx.x;
  const int tx = lin & 15, ty = lin >> 4;
  const int j0 = blockIdx.x * 16, i0 = blockIdx.y * 16;
  if (lin < 182) sw[lin] = weff[lin];

  float acc = 0.f;
  const bool interior = (i0 >= 1) && (i0 + 17 <= NN) && (j0 >= 1) && (j0 + 17 <= NN);
  if (interior) {
    for (int c = 0; c < CHREL; ++c) {
      const float* base = rel + (size_t)c * NN * NN + (size_t)(i0 - 1) * NN + (j0 - 1);
      {
        const int r = lin / 18, s = lin % 18;               // elems 0..255
        const float* gp = base + (size_t)r * NN + s;
        const unsigned la = (unsigned)(size_t)&st[r][s];    // low 32b = LDS offset
        asm volatile("global_load_async_to_lds_b32 %0, %1, off"
                     :: "v"(la), "v"(gp) : "memory");
      }
      if (lin < 68) {                                       // elems 256..323
        const int idx = 256 + lin;
        const int r = idx / 18, s = idx % 18;
        const float* gp = base + (size_t)r * NN + s;
        const unsigned la = (unsigned)(size_t)&st[r][s];
        asm volatile("global_load_async_to_lds_b32 %0, %1, off"
                     :: "v"(la), "v"(gp) : "memory");
      }
      asm volatile("s_wait_asynccnt 0x0" ::: "memory");
      __syncthreads();
#pragma unroll
      for (int r = 0; r < 3; ++r)
#pragma unroll
        for (int s2 = 0; s2 < 3; ++s2)
          acc += sw[c * 9 + r * 3 + s2] * st[ty + r][tx + s2];
      __syncthreads();
    }
  } else {
    for (int c = 0; c < CHREL; ++c) {
      for (int idx = lin; idx < 324; idx += 256) {
        const int r = idx / 18, s = idx % 18;
        const int ii = i0 - 1 + r, jj = j0 - 1 + s;
        float v = 0.f;
        if (ii >= 0 && ii < NN && jj >= 0 && jj < NN)
          v = rel[(size_t)c * NN * NN + (size_t)ii * NN + jj];
        st[r][s] = v;
      }
      __syncthreads();
#pragma unroll
      for (int r = 0; r < 3; ++r)
#pragma unroll
        for (int s2 = 0; s2 < 3; ++s2)
          acc += sw[c * 9 + r * 3 + s2] * st[ty + r][tx + s2];
      __syncthreads();
    }
  }
  g[(size_t)(i0 + ty) * NN + (j0 + tx)] = acc + sw[180];
}

// ---------------------------------------------------------------------------
// A[i,j] = ((g[i,j]+g[j,i])/2 + bdiff) > 0 ; store A^T (coalesced in i) and deg.
// ---------------------------------------------------------------------------
__global__ void __launch_bounds__(256) adj_deg_kernel(
    const float* __restrict__ g, const float* __restrict__ weff,
    float* __restrict__ AT, float* __restrict__ deg)
{
  const int tx = threadIdx.x & 15, ty = threadIdx.x >> 4;
  const int i = blockIdx.x * 16 + tx;
  const int j = blockIdx.y * 16 + ty;
  __shared__ float tile[16][17];
  const float bdiff = weff[181];
  const float gij = g[(size_t)i * NN + j];
  const float gji = g[(size_t)j * NN + i];
  const float a = (0.5f * (gij + gji) + bdiff) > 0.f ? 1.f : 0.f;
  AT[(size_t)j * NN + i] = a;
  tile[ty][tx] = a;
  __syncthreads();
  if (tx == 0) {
    float s = 0.f;
#pragma unroll
    for (int q = 0; q < 16; ++q) s += tile[ty][q];
    atomicAdd(deg + j, s);
  }
}

__global__ void zero_kernel(float* __restrict__ p, int n) {
  const int i = blockIdx.x * blockDim.x + threadIdx.x;
  if (i < n) p[i] = 0.f;
}
__global__ void invdeg_kernel(const float* __restrict__ deg, float* __restrict__ inv, int n) {
  const int i = blockIdx.x * blockDim.x + threadIdx.x;
  if (i < n) inv[i] = 1.0f / fmaxf(deg[i], 1.0f);
}

// cat[:,1024:1536] = bank_emb[bank_ids]
__global__ void bemb_kernel(const int* __restrict__ ids, const float* __restrict__ emb,
                            float* __restrict__ cat) {
  const int idx = blockIdx.x * blockDim.x + threadIdx.x;  // n*512+d
  const int n = idx >> 9, d = idx & 511;
  cat[(size_t)n * 1536 + 1024 + d] = emb[(size_t)ids[n] * 512 + d];
}

// ---------------------------------------------------------------------------
// Orchestration. Param leaf order = setup_inputs() insertion order with the
// nested params dict flattened jax-style (dict keys sorted, lists in order):
//  0 x_feat, 1 rel, 2 char_feats, 3 bank_ids,
//  4 bank_emb, 5 gb_c0_W, 6 gb_c0_b, 7 gb_c1_W, 8 gb_c1_b, 9 gb_s_W, 10 gb_s_b,
//  11.. layers[l] (20 leaves: en_b,en_g,et_W,et_b,et_ln_b,et_ln_g,
//                  g0{Wl,Wr,bl,ln_b,ln_g}, g1{...}, post_b,post_g,pre_b,pre_g),
//  91 Whh_b, 92 Whh_f, 93 Wih_b, 94 Wih_f, 95 b_b, 96 b_f,
//  97 proj_W, 98 proj_b, 99 xproj_W, 100 xproj_b
// edge_attr path is dead code w.r.t. the output -> eliminated.
// ---------------------------------------------------------------------------
extern "C" void kernel_launch(void* const* d_in, const int* in_sizes, int n_in,
                              void* d_out, int out_size, void* d_ws, size_t ws_size,
                              hipStream_t stream) {
  (void)in_sizes; (void)n_in; (void)out_size; (void)ws_size;
  const float* x_feat   = (const float*)d_in[0];
  const float* rel      = (const float*)d_in[1];
  const float* xchar    = (const float*)d_in[2];
  const int*   bank     = (const int*)d_in[3];
  const float* bank_emb = (const float*)d_in[4];
  const float* c0W = (const float*)d_in[5];
  const float* c0b = (const float*)d_in[6];
  const float* c1W = (const float*)d_in[7];
  const float* c1b = (const float*)d_in[8];
  const float* sW  = (const float*)d_in[9];
  const float* sb  = (const float*)d_in[10];
  auto LP = [&](int l, int off) { return (const float*)d_in[11 + l * 20 + off]; };
  const float* Whh_b = (const float*)d_in[91];
  const float* Whh_f = (const float*)d_in[92];
  const float* Wih_b = (const float*)d_in[93];
  const float* Wih_f = (const float*)d_in[94];
  const float* b_b   = (const float*)d_in[95];
  const float* b_f   = (const float*)d_in[96];
  const float* proj_W  = (const float*)d_in[97];
  const float* proj_b  = (const float*)d_in[98];
  const float* xproj_W = (const float*)d_in[99];
  const float* xproj_b = (const float*)d_in[100];

  char* ws = (char*)d_ws;
  float* cat    = (float*)(ws + 0);          // [1024,1536]  6,291,456 B
  float* Xbuf   = (float*)(ws + 6291456);    // [1024,512]
  float* Hbuf   = (float*)(ws + 8388608);    // [1024,512]   gelu(LN(x))
  float* YS     = (float*)(ws + 10485760);   // [1024,512]   group outputs
  float* HR     = (float*)(ws + 12582912);   // [1024,256]   relu(LN(y_in))
  float* T1     = (float*)(ws + 13631488);   // [1024,256]   scaled aggregation
  float* gmap   = (float*)(ws + 14680064);   // [1024,1024]
  float* AT     = (float*)(ws + 18874368);   // [1024,1024]
  float* deg    = (float*)(ws + 23068672);   // [1024]
  float* invdeg = (float*)(ws + 23072768);   // [1024]
  float* weff   = (float*)(ws + 23076864);   // [182]

  // --- graph builder (folded) ---
  prep_weff_kernel<<<1, 192, 0, stream>>>(c0W, c0b, c1W, c1b, sW, sb, weff);
  conv_gb_kernel<<<dim3(64, 64), 256, 0, stream>>>(rel, weff, gmap);
  zero_kernel<<<4, 256, 0, stream>>>(deg, NN);
  adj_deg_kernel<<<dim3(64, 64), 256, 0, stream>>>(gmap, weff, AT, deg);
  invdeg_kernel<<<4, 256, 0, stream>>>(deg, invdeg, NN);

  // --- node embedding: cat = [xproj | bilstm hsum | bank_emb gather] ---
  bilstm_kernel<<<dim3(64, 2), 256, 0, stream>>>(xchar, Wih_f, Whh_f, b_f,
                                                 Wih_b, Whh_b, b_b, cat);
  bemb_kernel<<<2048, 256, 0, stream>>>(bank, bank_emb, cat);
  gemm_wmma_kernel<<<dim3(32, 64), 32, 0, stream>>>(   // xp -> cat[:,0:512]
      x_feat, 768, xproj_W, 768, 0, xproj_b, nullptr, nullptr, 0,
      cat, 1536, NN, 512, 768, 0);
  gemm_wmma_kernel<<<dim3(32, 64), 32, 0, stream>>>(   // x = cat @ proj_W^T + b
      cat, 1536, proj_W, 1536, 0, proj_b, nullptr, nullptr, 0,
      Xbuf, 512, NN, 512, 1536, 0);

  // --- reversible group SAGE layers ---
  for (int l = 0; l < 4; ++l) {
    // h = gelu(LN(x; pre))
    ln_kernel<<<NN, 256, 0, stream>>>(Xbuf, 512, nullptr, 0,
                                      LP(l, 19), LP(l, 18), Hbuf, 512, DM, 2);
    for (int gi = 0; gi < 2; ++gi) {
      const float* yin = (gi == 0) ? (Hbuf + DH) : YS;          // y_{-1}=x_{G-1}
      const float* Wl  = LP(l, 6 + gi * 5 + 0);
      const float* Wr  = LP(l, 6 + gi * 5 + 1);
      const float* bl  = LP(l, 6 + gi * 5 + 2);
      const float* lnb = LP(l, 6 + gi * 5 + 3);
      const float* lng = LP(l, 6 + gi * 5 + 4);
      // HR = relu(LN(y_in))
      ln_kernel<<<NN, 256, 0, stream>>>(yin, 512, nullptr, 0, lng, lnb,
                                        HR, DH, DH, 1);
      // T1 = (A^T @ HR) / max(deg,1)     (B is K-major [1024,256])
      gemm_wmma_kernel<<<dim3(16, 64), 32, 0, stream>>>(
          AT, 1024, HR, 256, 1, nullptr, invdeg, nullptr, 0,
          T1, 256, NN, DH, 1024, 0);
      // y = xg + T1 @ Wl^T + bl ; then y += HR @ Wr^T
      gemm_wmma_kernel<<<dim3(16, 64), 32, 0, stream>>>(
          T1, 256, Wl, 256, 0, bl, nullptr, Hbuf + gi * DH, 512,
          YS + gi * DH, 512, NN, DH, 256, 0);
      gemm_wmma_kernel<<<dim3(16, 64), 32, 0, stream>>>(
          HR, 256, Wr, 256, 0, nullptr, nullptr, nullptr, 0,
          YS + gi * DH, 512, NN, DH, 256, 1);
    }
    // x = LN(ys + x_res; post); last layer fuses final gelu -> d_out
    if (l < 3)
      ln_kernel<<<NN, 256, 0, stream>>>(YS, 512, Xbuf, 512,
                                        LP(l, 17), LP(l, 16), Xbuf, 512, DM, 0);
    else
      ln_kernel<<<NN, 256, 0, stream>>>(YS, 512, Xbuf, 512,
                                        LP(l, 17), LP(l, 16),
                                        (float*)d_out, 512, DM, 2);
  }
}
